// BinaryDense_87282325389674
// MI455X (gfx1250) — matrix operand order repaired and checked by reading, not compiled, not run on hardware
//
#include <hip/hip_runtime.h>
#include <stdint.h>

// Problem dims (fixed by the reference)
#define BDIM   4096
#define DIN    4096
#define NUNITS 4096

typedef _Float16 half_t;
typedef __attribute__((ext_vector_type(16))) _Float16 v16h;
typedef __attribute__((ext_vector_type(8)))  _Float16 v8h;
typedef __attribute__((ext_vector_type(8)))  float    v8f;
typedef __attribute__((ext_vector_type(4)))  float    v4f;

union AFrag { v16h v; v8h h[2]; };
union Acc   { v8f  v; float f[8]; };

// ---------------------------------------------------------------------------
// CDNA5 async global->LDS copy helpers (ASYNCcnt-tracked DMA, no VGPR transit)
// ---------------------------------------------------------------------------
__device__ __forceinline__ void async_copy_b128(uint32_t lds_addr, const void* gaddr) {
    asm volatile("global_load_async_to_lds_b128 %0, %1, off"
                 :: "v"(lds_addr), "v"(gaddr) : "memory");
}

__device__ __forceinline__ void async_wait0() {
#if __has_builtin(__builtin_amdgcn_s_wait_asynccnt)
    __builtin_amdgcn_s_wait_asynccnt(0);
#else
    asm volatile("s_wait_asynccnt 0x0" ::: "memory");
#endif
}

// ---------------------------------------------------------------------------
// Pass 1: per-block partial sums of |w| (deterministic tree reduction)
// ---------------------------------------------------------------------------
__global__ __launch_bounds__(256)
void absmean_partial(const float* __restrict__ w, float* __restrict__ partial, int n) {
    __shared__ float red[256];
    const int tid = threadIdx.x;
    int idx = blockIdx.x * 256 + tid;
    const int stride = gridDim.x * 256;
    float s = 0.0f;
    for (int i = idx; i < n; i += stride) s += fabsf(w[i]);
    red[tid] = s;
    __syncthreads();
    #pragma unroll
    for (int off = 128; off > 0; off >>= 1) {
        if (tid < off) red[tid] += red[tid + off];
        __syncthreads();
    }
    if (tid == 0) partial[blockIdx.x] = red[0];
}

// ---------------------------------------------------------------------------
// Pass 2: reduce 1024 partials -> threshold = mean(|w|) at thr[0]
// ---------------------------------------------------------------------------
__global__ __launch_bounds__(256)
void absmean_final(const float* __restrict__ partial, float* __restrict__ thr) {
    __shared__ float red[256];
    const int tid = threadIdx.x;
    float s = 0.0f;
    #pragma unroll
    for (int i = 0; i < 4; ++i) s += partial[tid + i * 256];
    red[tid] = s;
    __syncthreads();
    #pragma unroll
    for (int off = 128; off > 0; off >>= 1) {
        if (tid < off) red[tid] += red[tid + off];
        __syncthreads();
    }
    if (tid == 0) thr[0] = red[0] * (1.0f / 16777216.0f);
}

// ---------------------------------------------------------------------------
// Pre-pass A: convert inputs f32 -> f16 once (A is reused by 32 N-tiles)
// ---------------------------------------------------------------------------
__global__ __launch_bounds__(256)
void convert_a_f16(const float* __restrict__ A, half_t* __restrict__ Ah, int n8) {
    int i = blockIdx.x * 256 + threadIdx.x;
    const int stride = gridDim.x * 256;
    for (; i < n8; i += stride) {
        const float* s = A + (size_t)i * 8;
        v4f f0 = *(const v4f*)(s);
        v4f f1 = *(const v4f*)(s + 4);
        v8h h;
        #pragma unroll
        for (int j = 0; j < 4; ++j) { h[j] = (half_t)f0[j]; h[4 + j] = (half_t)f1[j]; }
        *(v8h*)(Ah + (size_t)i * 8) = h;
    }
}

// ---------------------------------------------------------------------------
// Pre-pass B: quantize W and store TRANSPOSED f16: WqT[n][k]  (64x64 LDS tiles)
// ---------------------------------------------------------------------------
__global__ __launch_bounds__(256)
void quant_transpose_w(const float* __restrict__ W, const float* __restrict__ thrp,
                       half_t* __restrict__ WqT) {
    __shared__ half_t q[64][72];           // [n][k], padded
    const int tid = threadIdx.x;
    const int n0 = blockIdx.x * 64;
    const int k0 = blockIdx.y * 64;
    const float thr = thrp[0];

    // Phase 1: coalesced read of two k-rows x 8 n, quantize, transposed LDS write
    const int kp = tid >> 3;               // 0..31 -> k = 2*kp, 2*kp+1
    const int nc = tid & 7;                // 0..7  -> n = nc*8 .. +7
    {
        const float* s0 = W + (size_t)(k0 + 2 * kp) * NUNITS + n0 + nc * 8;
        const float* s1 = s0 + NUNITS;
        v4f w00 = *(const v4f*)(s0);
        v4f w01 = *(const v4f*)(s0 + 4);
        v4f w10 = *(const v4f*)(s1);
        v4f w11 = *(const v4f*)(s1 + 4);
        #pragma unroll
        for (int j = 0; j < 8; ++j) {
            float a0 = (j < 4) ? w00[j] : w01[j - 4];
            float a1 = (j < 4) ? w10[j] : w11[j - 4];
            half_t* dst = &q[nc * 8 + j][2 * kp];
            dst[0] = (fabsf(a0) > thr) ? (half_t)1.0f : (half_t)0.0f;
            dst[1] = (fabsf(a1) > thr) ? (half_t)1.0f : (half_t)0.0f;
        }
    }
    __syncthreads();

    // Phase 2: coalesced b128 writes of transposed tile
    const int nr = tid >> 2;               // 0..63
    const int kc = (tid & 3) * 16;         // 0,16,32,48
    v8h h0 = *(const v8h*)&q[nr][kc];
    v8h h1 = *(const v8h*)&q[nr][kc + 8];
    half_t* out = WqT + (size_t)(n0 + nr) * DIN + k0 + kc;
    *(v8h*)(out)     = h0;
    *(v8h*)(out + 8) = h1;
}

// ---------------------------------------------------------------------------
// Shared WMMA micro-tile: 4x2 tiles of 16x16, K=32 slice at kofs
// ---------------------------------------------------------------------------
constexpr int BM = 128, BN = 128;

template <int LDTW>
__device__ __forceinline__
void tile_mma(const half_t (* __restrict__ As)[LDTW], const half_t (* __restrict__ Bs)[LDTW],
              int wm, int wn, int frow, int kb_a, int kb_b, int kofs, Acc acc[4][2]) {
    AFrag af[4], bf[2];
    #pragma unroll
    for (int mt = 0; mt < 4; ++mt) {
        const half_t* ap = &As[wm * 64 + mt * 16 + frow][kofs];
        af[mt].h[0] = *(const v8h*)(ap + kb_a);
        af[mt].h[1] = *(const v8h*)(ap + kb_a + 8);
    }
    #pragma unroll
    for (int nt = 0; nt < 2; ++nt) {
        const half_t* bp = &Bs[wn * 32 + nt * 16 + frow][kofs];
        bf[nt].h[0] = *(const v8h*)(bp + kb_b);
        bf[nt].h[1] = *(const v8h*)(bp + kb_b + 8);
    }
    #pragma unroll
    for (int mt = 0; mt < 4; ++mt)
        #pragma unroll
        for (int nt = 0; nt < 2; ++nt)
            acc[mt][nt].v = __builtin_amdgcn_wmma_f32_16x16x32_f16(
                false, af[mt].v, false, bf[nt].v,
                (short)0, acc[mt][nt].v, false, false);
}

__device__ __forceinline__
void store_output(Acc acc[4][2], const float* __restrict__ bias, float* __restrict__ C,
                  int m0, int n0, int wm, int wn, int lane) {
    const int col  = lane & 15;
    const int roff = (lane >> 4) * 8;
    #pragma unroll
    for (int nt = 0; nt < 2; ++nt) {
        const int n  = n0 + wn * 32 + nt * 16 + col;
        const float bv = bias[n];
        #pragma unroll
        for (int mt = 0; mt < 4; ++mt) {
            const int mb = m0 + wm * 64 + mt * 16 + roff;
            float* cp = C + (size_t)mb * NUNITS + n;
            #pragma unroll
            for (int rr = 0; rr < 8; ++rr)
                cp[(size_t)rr * NUNITS] = acc[mt][nt].f[rr] + bv;
        }
    }
}

// ---------------------------------------------------------------------------
// FAST GEMM (pre-converted f16 inputs): BK=64, double-buffered LDS,
// 16 WMMAs per barrier, staging via async global->LDS DMA (ASYNCcnt).
// ---------------------------------------------------------------------------
constexpr int FBK = 64, FLDT = 72;

__global__ __launch_bounds__(256)
void binary_dense_wmma_fast(const half_t* __restrict__ Ah,
                            const half_t* __restrict__ WqT,
                            const float* __restrict__ bias,
                            float* __restrict__ C) {
    __shared__ half_t As[2][BM][FLDT];
    __shared__ half_t Bs[2][BN][FLDT];

    const int tid  = threadIdx.x;
    const int lane = tid & 31;
    const int wave = tid >> 5;
    const int wm   = wave & 1;
    const int wn   = wave >> 1;
    const int m0   = blockIdx.y * BM;
    const int n0   = blockIdx.x * BN;

    const int s_row = tid >> 1;            // 0..127
    const int s_col = (tid & 1) * 32;      // 0 or 32

    const int frow = lane & 15;
    const int kb_a = (lane >> 4) * 16;
    const int kb_b = (lane >> 4) * 16;

    Acc acc[4][2];
    v8f zero = {};
    #pragma unroll
    for (int i = 0; i < 4; ++i)
        #pragma unroll
        for (int j = 0; j < 2; ++j) acc[i][j].v = zero;

    // Async DMA: 4x b128 for A slab + 4x b128 for B slab per thread
    auto async_stage = [&](int k0, int buf) {
        const half_t* ap = Ah  + (size_t)(m0 + s_row) * DIN + k0 + s_col;
        const half_t* bp = WqT + (size_t)(n0 + s_row) * DIN + k0 + s_col;
        const uint32_t la = (uint32_t)(uintptr_t)&As[buf][s_row][s_col];
        const uint32_t lb = (uint32_t)(uintptr_t)&Bs[buf][s_row][s_col];
        #pragma unroll
        for (int i = 0; i < 4; ++i) {
            async_copy_b128(la + 16 * i, ap + 8 * i);
            async_copy_b128(lb + 16 * i, bp + 8 * i);
        }
    };

    async_stage(0, 0);
    async_wait0();
    __syncthreads();

    constexpr int ITERS = DIN / FBK;       // 64
    for (int kt = 0; kt < ITERS - 1; ++kt) {
        const int cur = kt & 1;
        const int nxt = cur ^ 1;
        async_stage((kt + 1) * FBK, nxt);  // DMA next tile under the WMMAs
        tile_mma<FLDT>(As[cur], Bs[cur], wm, wn, frow, kb_a, kb_b, 0,  acc);
        tile_mma<FLDT>(As[cur], Bs[cur], wm, wn, frow, kb_a, kb_b, 32, acc);
        async_wait0();                     // own DMA into nxt complete
        __syncthreads();                   // visible to all waves
    }
    const int last = (ITERS - 1) & 1;
    tile_mma<FLDT>(As[last], Bs[last], wm, wn, frow, kb_a, kb_b, 0,  acc);
    tile_mma<FLDT>(As[last], Bs[last], wm, wn, frow, kb_a, kb_b, 32, acc);

    store_output(acc, bias, C, m0, n0, wm, wn, lane);
}

// ---------------------------------------------------------------------------
// FALLBACK GEMM (small d_ws): inline f32->f16 convert + quantize, BK=32
// ---------------------------------------------------------------------------
constexpr int BK = 32, LDT = 40;

struct StageRegs { v4f a0, a1, a2, a3, w00, w01, w10, w11; };

__device__ __forceinline__
void stage_load(const float* __restrict__ A, const float* __restrict__ W,
                int m0, int n0, int k0,
                int a_row, int a_col, int b_kp, int b_nc, StageRegs& r) {
    const float* src = A + (size_t)(m0 + a_row) * DIN + k0 + a_col;
    r.a0 = *(const v4f*)(src);
    r.a1 = *(const v4f*)(src + 4);
    r.a2 = *(const v4f*)(src + 8);
    r.a3 = *(const v4f*)(src + 12);
    const float* s0 = W + (size_t)(k0 + 2 * b_kp) * NUNITS + n0 + b_nc * 8;
    const float* s1 = s0 + NUNITS;
    r.w00 = *(const v4f*)(s0);
    r.w01 = *(const v4f*)(s0 + 4);
    r.w10 = *(const v4f*)(s1);
    r.w11 = *(const v4f*)(s1 + 4);
}

__device__ __forceinline__
void stage_store(const StageRegs& r, float thr,
                 int a_row, int a_col, int b_kp, int b_nc,
                 half_t (* __restrict__ As)[LDT], half_t (* __restrict__ Bs)[LDT]) {
    v8h h0, h1;
    #pragma unroll
    for (int i = 0; i < 4; ++i) {
        h0[i]     = (half_t)r.a0[i];
        h0[4 + i] = (half_t)r.a1[i];
        h1[i]     = (half_t)r.a2[i];
        h1[4 + i] = (half_t)r.a3[i];
    }
    *(v8h*)&As[a_row][a_col]     = h0;
    *(v8h*)&As[a_row][a_col + 8] = h1;
    #pragma unroll
    for (int j = 0; j < 8; ++j) {
        float a0 = (j < 4) ? r.w00[j] : r.w01[j - 4];
        float a1 = (j < 4) ? r.w10[j] : r.w11[j - 4];
        half_t* dst = &Bs[b_nc * 8 + j][2 * b_kp];
        dst[0] = (fabsf(a0) > thr) ? (half_t)1.0f : (half_t)0.0f;
        dst[1] = (fabsf(a1) > thr) ? (half_t)1.0f : (half_t)0.0f;
    }
}

__global__ __launch_bounds__(256)
void binary_dense_wmma(const float* __restrict__ A,
                       const float* __restrict__ W,
                       const float* __restrict__ bias,
                       const float* __restrict__ thrp,
                       float* __restrict__ C) {
    __shared__ half_t As[2][BM][LDT];
    __shared__ half_t Bs[2][BN][LDT];

    const int tid  = threadIdx.x;
    const int lane = tid & 31;
    const int wave = tid >> 5;
    const int wm   = wave & 1;
    const int wn   = wave >> 1;
    const int m0   = blockIdx.y * BM;
    const int n0   = blockIdx.x * BN;
    const float thr = thrp[0];

    const int a_row = tid >> 1;
    const int a_col = (tid & 1) * 16;
    const int b_kp  = tid >> 4;
    const int b_nc  = tid & 15;

    const int frow = lane & 15;
    const int kb_a = (lane >> 4) * 16;
    const int kb_b = (lane >> 4) * 16;

    Acc acc[4][2];
    v8f zero = {};
    #pragma unroll
    for (int i = 0; i < 4; ++i)
        #pragma unroll
        for (int j = 0; j < 2; ++j) acc[i][j].v = zero;

    StageRegs r;
    stage_load(A, W, m0, n0, 0, a_row, a_col, b_kp, b_nc, r);
    stage_store(r, thr, a_row, a_col, b_kp, b_nc, As[0], Bs[0]);
    __syncthreads();

    constexpr int ITERS = DIN / BK;        // 128
    for (int kt = 0; kt < ITERS - 1; ++kt) {
        const int cur = kt & 1;
        const int nxt = cur ^ 1;
        stage_load(A, W, m0, n0, (kt + 1) * BK, a_row, a_col, b_kp, b_nc, r);
        tile_mma<LDT>(As[cur], Bs[cur], wm, wn, frow, kb_a, kb_b, 0, acc);
        stage_store(r, thr, a_row, a_col, b_kp, b_nc, As[nxt], Bs[nxt]);
        __syncthreads();
    }
    const int last = (ITERS - 1) & 1;
    tile_mma<LDT>(As[last], Bs[last], wm, wn, frow, kb_a, kb_b, 0, acc);

    store_output(acc, bias, C, m0, n0, wm, wn, lane);
}

// ---------------------------------------------------------------------------
extern "C" void kernel_launch(void* const* d_in, const int* in_sizes, int n_in,
                              void* d_out, int out_size, void* d_ws, size_t ws_size,
                              hipStream_t stream) {
    const float* A    = (const float*)d_in[0];   // inputs [B, D_IN]
    const float* W    = (const float*)d_in[1];   // kernel [D_IN, UNITS]
    const float* bias = (const float*)d_in[2];   // [UNITS]
    float* out = (float*)d_out;

    char*  ws      = (char*)d_ws;
    float* partial = (float*)ws;                       // 1024 floats
    float* thr     = (float*)(ws + 4096);              // 1 float

    const size_t HALF_BUF = (size_t)DIN * NUNITS * sizeof(half_t);  // 32 MB
    const size_t NEED     = 8192 + 2 * HALF_BUF;

    absmean_partial<<<1024, 256, 0, stream>>>(W, partial, DIN * NUNITS);
    absmean_final<<<1, 256, 0, stream>>>(partial, thr);

    dim3 grid(NUNITS / 128, BDIM / 128);
    if (ws_size >= NEED) {
        half_t* Ah  = (half_t*)(ws + 8192);
        half_t* WqT = (half_t*)(ws + 8192 + HALF_BUF);
        convert_a_f16<<<4096, 256, 0, stream>>>(A, Ah, (BDIM * DIN) / 8);
        dim3 qgrid(NUNITS / 64, DIN / 64);
        quant_transpose_w<<<qgrid, 256, 0, stream>>>(W, thr, WqT);
        binary_dense_wmma_fast<<<grid, 256, 0, stream>>>(Ah, WqT, bias, out);
    } else {
        binary_dense_wmma<<<grid, 256, 0, stream>>>(A, W, bias, thr, out);
    }
}
